// KAN_ActorNetwork_27496380629351
// MI455X (gfx1250) — matrix operand order repaired and checked by reading, not compiled, not run on hardware
//
#include <hip/hip_runtime.h>
#include <hip/hip_bf16.h>

typedef __attribute__((ext_vector_type(16))) _Float16 v16h;
typedef __attribute__((ext_vector_type(8)))  _Float16 v8h;
typedef __attribute__((ext_vector_type(8)))  float    v8f;

#define B_TOTAL 65536
#define D_IN    128
#define HID1    64
#define HID2    32
#define D_OUT   16
#define GEN     9                 // features per input: silu + 8 spline bases
#define K1      (D_IN*GEN)        // 1152
#define K2      (HID1*GEN)        // 576
#define K3      (HID2*GEN)        // 288
#define M_TILE  64
#define BLOCK   128               // 4 wave32
#define CH_STRIDE 296             // 288 + 8 pad (f16 elems), 16B aligned, bank-conflict free
#define H_OFF   37888             // byte offset of h/LN buffer (64*296*2)
#define H1_STRIDE 65              // f32 stride pad -> distinct banks per row
#define H2_STRIDE 33
#define LDS_BYTES (H_OFF + M_TILE*H1_STRIDE*4)   // 37888 + 16640 = 54528

__device__ __forceinline__ v8f zero8() {
  v8f z = {0.f,0.f,0.f,0.f,0.f,0.f,0.f,0.f};
  return z;
}

__device__ __forceinline__ float silu_f(float x) {
  return x * (1.0f / (1.0f + __expf(-x)));
}

// Analytic uniform cubic B-spline feature writer.
// Grid: knots t_j = 0.4*(j-3) - 1, j=0..11 (uniform incl. ghost knots), so the
// Cox-de Boor recursion of the reference collapses to the classic uniform
// cubic pieces: for x in [t_j, t_j+1), u = frac, the only nonzero bases are
// N_{j-3..j} = { (1-u)^3, 3u^3-6u^2+4, -3u^3+3u^2+3u+1, u^3 } / 6.
// Caller guarantees dst[1..8] are pre-zeroed.
__device__ __forceinline__ void write_feats(_Float16* dst, float x) {
  dst[0] = (_Float16)silu_f(x);
  float xc = (x + 2.2f) * 2.5f;            // (x - t_0) / h
  if (xc >= 0.0f && xc < 11.0f) {
    int   j  = (int)xc;                    // order-0 interval, 0..10
    float u  = xc - (float)j;
    float v  = 1.0f - u;
    float u2 = u * u;
    float w0 = v * v * v * (1.0f / 6.0f);
    float w1 = (0.5f * u - 1.0f) * u2 + (2.0f / 3.0f);
    float w2 = ((-0.5f * u + 0.5f) * u + 0.5f) * u + (1.0f / 6.0f);
    float w3 = u * u2 * (1.0f / 6.0f);
    int lo = j - 3;
    if (lo + 0 >= 0 && lo + 0 < 8) dst[1 + lo + 0] = (_Float16)w0;
    if (lo + 1 >= 0 && lo + 1 < 8) dst[1 + lo + 1] = (_Float16)w1;
    if (lo + 2 >= 0 && lo + 2 < 8) dst[1 + lo + 2] = (_Float16)w2;
    if (lo + 3 >= 0 && lo + 3 < 8) dst[1 + lo + 3] = (_Float16)w3;
  }
}

// One K-chunk of WMMA: A (16x32 f16 per wave) from LDS, B (32x16 f16) from global
// (L2-resident packed weights). Fragment layouts used here (contiguous k, same order in A and B):
//  A: lane m (0-15) K=[kb+16hi, +16) contiguous in v0-7
//  B: lane n (0-15) col n, K=[kb+16hi, +16) contiguous (32B aligned)
template <int NSLICES, int NT, int KW>
__device__ __forceinline__ void wmma_chunk(const _Float16* __restrict__ chunkBuf,
                                           const _Float16* __restrict__ W,
                                           int ksBase, v8f acc[NT]) {
  const int lane = threadIdx.x & 31;
  const int wave = threadIdx.x >> 5;
  const int m    = lane & 15;
  const int hi   = lane >> 4;
  const _Float16* arow = chunkBuf + (wave * 16 + m) * CH_STRIDE + hi * 16;
#pragma unroll
  for (int ks = 0; ks < NSLICES; ++ks) {
    v8h a0 = *(const v8h*)(arow + ks * 32);
    v8h a1 = *(const v8h*)(arow + ks * 32 + 8);
    v16h A = __builtin_shufflevector(a0, a1, 0,1,2,3,4,5,6,7,8,9,10,11,12,13,14,15);
#pragma unroll
    for (int nt = 0; nt < NT; ++nt) {
      const _Float16* wp = W + (nt * 16 + m) * KW + (ksBase + ks) * 32 + hi * 16;
      v16h Bf = *(const v16h*)wp;
      acc[nt] = __builtin_amdgcn_wmma_f32_16x16x32_f16(
          false, A, false, Bf, (short)0, acc[nt], false, false);
    }
  }
}

// Pack fp32 weights -> f16, K-order interleaved per input: [silu, b0..b7].
__global__ void pack_weights(const float* __restrict__ bw1, const float* __restrict__ sw1,
                             const float* __restrict__ bw2, const float* __restrict__ sw2,
                             const float* __restrict__ bw3, const float* __restrict__ sw3,
                             _Float16* __restrict__ ws) {
  const int N1 = HID1 * K1, N2 = HID2 * K2, N3 = D_OUT * K3;
  int idx = blockIdx.x * blockDim.x + threadIdx.x;
  if (idx < N1) {
    int o = idx / K1, k = idx - o * K1, i = k / GEN, r = k - i * GEN;
    float v = (r == 0) ? bw1[o * D_IN + i] : sw1[(o * D_IN + i) * 8 + (r - 1)];
    ws[idx] = (_Float16)v;
  } else if (idx < N1 + N2) {
    int t = idx - N1;
    int o = t / K2, k = t - o * K2, i = k / GEN, r = k - i * GEN;
    float v = (r == 0) ? bw2[o * HID1 + i] : sw2[(o * HID1 + i) * 8 + (r - 1)];
    ws[idx] = (_Float16)v;
  } else if (idx < N1 + N2 + N3) {
    int t = idx - (N1 + N2);
    int o = t / K3, k = t - o * K3, i = k / GEN, r = k - i * GEN;
    float v = (r == 0) ? bw3[o * HID2 + i] : sw3[(o * HID2 + i) * 8 + (r - 1)];
    ws[idx] = (_Float16)v;
  }
}

__global__ __launch_bounds__(BLOCK)
void kan_fused(const float* __restrict__ obs,
               const _Float16* __restrict__ Wt,
               const float* __restrict__ ln1_g, const float* __restrict__ ln1_b,
               const float* __restrict__ ln2_g, const float* __restrict__ ln2_b,
               float* __restrict__ out) {
  extern __shared__ char smem[];
  _Float16* chunkBuf = (_Float16*)smem;
  float* hbuf = (float*)(smem + H_OFF);
  const int tid = threadIdx.x;
  const int rowBase = blockIdx.x * M_TILE;
  const int lane = tid & 31, wave = tid >> 5, m = lane & 15, hi = lane >> 4;

  const _Float16* W1 = Wt;
  const _Float16* W2 = Wt + HID1 * K1;
  const _Float16* W3 = Wt + HID1 * K1 + HID2 * K2;

  const v8h hz = {(_Float16)0.f,(_Float16)0.f,(_Float16)0.f,(_Float16)0.f,
                  (_Float16)0.f,(_Float16)0.f,(_Float16)0.f,(_Float16)0.f};

  // ---------------- layer 1: (64 x 1152) x (1152 x 64) ----------------
  v8f acc1[4];
#pragma unroll
  for (int n = 0; n < 4; ++n) acc1[n] = zero8();
  for (int c = 0; c < 4; ++c) {              // 4 K-chunks of 288
    for (int z = tid; z < M_TILE * CH_STRIDE / 8; z += BLOCK)
      ((v8h*)chunkBuf)[z] = hz;              // bulk zero (b128 stores)
    __syncthreads();
    for (int e = tid; e < M_TILE * 32; e += BLOCK) {
      int row = e >> 5, ii = e & 31;
      float x = obs[(rowBase + row) * D_IN + 32 * c + ii];
      write_feats(chunkBuf + row * CH_STRIDE + ii * GEN, x);
    }
    __syncthreads();
    wmma_chunk<9, 4, K1>(chunkBuf, W1, c * 9, acc1);
    __syncthreads();
  }
  // spill h1 to LDS (C/D layout: acc[nt][v] = (M = v+8*hi, N = nt*16+m))
#pragma unroll
  for (int nt = 0; nt < 4; ++nt)
#pragma unroll
    for (int v = 0; v < 8; ++v)
      hbuf[(wave * 16 + v + 8 * hi) * H1_STRIDE + nt * 16 + m] = acc1[nt][v];
  __syncthreads();
  // LayerNorm 1 (in place, one row per thread)
  if (tid < M_TILE) {
    float* hr = hbuf + tid * H1_STRIDE;
    float mu = 0.f;
#pragma unroll
    for (int i = 0; i < HID1; ++i) mu += hr[i];
    mu *= (1.0f / HID1);
    float var = 0.f;
#pragma unroll
    for (int i = 0; i < HID1; ++i) { float d = hr[i] - mu; var += d * d; }
    var *= (1.0f / HID1);
    float rs = rsqrtf(var + 1e-5f);
#pragma unroll
    for (int i = 0; i < HID1; ++i) hr[i] = (hr[i] - mu) * rs * ln1_g[i] + ln1_b[i];
  }
  __syncthreads();

  // ---------------- layer 2: (64 x 576) x (576 x 32) ----------------
  v8f acc2[2];
#pragma unroll
  for (int n = 0; n < 2; ++n) acc2[n] = zero8();
  for (int c = 0; c < 2; ++c) {
    for (int z = tid; z < M_TILE * CH_STRIDE / 8; z += BLOCK)
      ((v8h*)chunkBuf)[z] = hz;
    __syncthreads();
    for (int e = tid; e < M_TILE * 32; e += BLOCK) {
      int row = e >> 5, ii = e & 31;
      float x = hbuf[row * H1_STRIDE + 32 * c + ii];
      write_feats(chunkBuf + row * CH_STRIDE + ii * GEN, x);
    }
    __syncthreads();
    wmma_chunk<9, 2, K2>(chunkBuf, W2, c * 9, acc2);
    __syncthreads();
  }
#pragma unroll
  for (int nt = 0; nt < 2; ++nt)
#pragma unroll
    for (int v = 0; v < 8; ++v)
      hbuf[(wave * 16 + v + 8 * hi) * H2_STRIDE + nt * 16 + m] = acc2[nt][v];
  __syncthreads();
  // LayerNorm 2
  if (tid < M_TILE) {
    float* hr = hbuf + tid * H2_STRIDE;
    float mu = 0.f;
#pragma unroll
    for (int i = 0; i < HID2; ++i) mu += hr[i];
    mu *= (1.0f / HID2);
    float var = 0.f;
#pragma unroll
    for (int i = 0; i < HID2; ++i) { float d = hr[i] - mu; var += d * d; }
    var *= (1.0f / HID2);
    float rs = rsqrtf(var + 1e-5f);
#pragma unroll
    for (int i = 0; i < HID2; ++i) hr[i] = (hr[i] - mu) * rs * ln2_g[i] + ln2_b[i];
  }
  __syncthreads();

  // ---------------- layer 3: (64 x 288) x (288 x 16) -> logits ----------------
  v8f acc3 = zero8();
  for (int z = tid; z < M_TILE * CH_STRIDE / 8; z += BLOCK)
    ((v8h*)chunkBuf)[z] = hz;
  __syncthreads();
  for (int e = tid; e < M_TILE * 32; e += BLOCK) {
    int row = e >> 5, ii = e & 31;
    float x = hbuf[row * H2_STRIDE + ii];
    write_feats(chunkBuf + row * CH_STRIDE + ii * GEN, x);
  }
  __syncthreads();
  {
    v8f a3[1] = {acc3};
    wmma_chunk<9, 1, K3>(chunkBuf, W3, 0, a3);
    acc3 = a3[0];
  }
#pragma unroll
  for (int v = 0; v < 8; ++v)
    out[(rowBase + wave * 16 + v + 8 * hi) * D_OUT + m] = acc3[v];
}

extern "C" void kernel_launch(void* const* d_in, const int* in_sizes, int n_in,
                              void* d_out, int out_size, void* d_ws, size_t ws_size,
                              hipStream_t stream) {
  (void)in_sizes; (void)n_in; (void)out_size; (void)ws_size;
  const float* obs = (const float*)d_in[0];
  // d_in[1], d_in[6], d_in[11] are the grids (uniform, recomputed analytically)
  const float* bw1 = (const float*)d_in[2];
  const float* sw1 = (const float*)d_in[3];
  const float* g1  = (const float*)d_in[4];
  const float* b1  = (const float*)d_in[5];
  const float* bw2 = (const float*)d_in[7];
  const float* sw2 = (const float*)d_in[8];
  const float* g2  = (const float*)d_in[9];
  const float* b2  = (const float*)d_in[10];
  const float* bw3 = (const float*)d_in[12];
  const float* sw3 = (const float*)d_in[13];
  _Float16* ws = (_Float16*)d_ws;
  float* out = (float*)d_out;

  const int NPACK = HID1 * K1 + HID2 * K2 + D_OUT * K3;   // 96768
  pack_weights<<<(NPACK + 255) / 256, 256, 0, stream>>>(bw1, sw1, bw2, sw2, bw3, sw3, ws);
  kan_fused<<<B_TOTAL / M_TILE, BLOCK, LDS_BYTES, stream>>>(obs, ws, g1, b1, g2, b2, out);
}
